// SCC_Conv2d_38405597560851
// MI455X (gfx1250) — matrix operand order repaired and checked by reading, not compiled, not run on hardware
//
#include <hip/hip_runtime.h>

typedef __attribute__((ext_vector_type(16))) __bf16 v16bf;
typedef __attribute__((ext_vector_type(8)))  float  v8f;
typedef __attribute__((ext_vector_type(4)))  int    v4i;

#define B_    32
#define OC_   256
#define IC_   256
#define H_    56
#define W_    56
#define HW_   3136      // 56*56
#define NUM_  64
#define KTOT_ 2304      // 9 * 256, K order: (kh*3+kw)*256 + ic
#define NWEIGHT_ 589824 // OC*IC*9

static __device__ __forceinline__ unsigned short f2bf(float f) {
    unsigned u = __builtin_bit_cast(unsigned, f);
    u += 0x7FFFu + ((u >> 16) & 1u);   // round-to-nearest-even
    return (unsigned short)(u >> 16);
}

// ---------------------------------------------------------------------------
// Kernel 1: per-sample weight synthesis, fp32 -> bf16, laid out as
// ws[b][oc][k], k = (kh*3+kw)*256 + ic.  Reads each W row once (151 MB),
// reuses it for all 32 batches -> pure HBM-bandwidth bound (~8 us).
// ---------------------------------------------------------------------------
__global__ __launch_bounds__(256)
void synth_w_kernel(const float* __restrict__ se, const float* __restrict__ Wt,
                    unsigned short* __restrict__ ws) {
    __shared__ float se_s[B_ * NUM_];
    const int t = threadIdx.x;
    for (int i = t; i < B_ * NUM_; i += 256) se_s[i] = se[i];
    __syncthreads();

    const long o = (long)blockIdx.x * 256 + t;   // 0..NWEIGHT_-1 (exact grid)
    if (o >= NWEIGHT_) return;
    const int oc  = (int)(o / (IC_ * 9));
    const int rem = (int)(o % (IC_ * 9));
    const int ic  = rem / 9;
    const int r   = rem % 9;                     // kh*3+kw

    float wrow[NUM_];
    const float* wp = Wt + o * NUM_;
    #pragma unroll
    for (int n = 0; n < NUM_; ++n) wrow[n] = wp[n];

    const long outbase = ((long)oc * 9 + r) * IC_ + ic;  // oc*KTOT_ + r*256 + ic
    for (int b = 0; b < B_; ++b) {
        const float* sb = &se_s[b * NUM_];
        float acc = 0.f;
        #pragma unroll
        for (int n = 0; n < NUM_; ++n) acc += sb[n] * wrow[n];
        ws[(long)b * ((long)OC_ * KTOT_) + outbase] = f2bf(acc);
    }
}

// ---------------------------------------------------------------------------
// Kernel 2: activations fp32 -> bf16
// ---------------------------------------------------------------------------
__global__ __launch_bounds__(256)
void cvt_x_kernel(const float* __restrict__ x, unsigned short* __restrict__ xb) {
    const long i0 = ((long)blockIdx.x * 256 + threadIdx.x) * 4;
    const float4 v = *(const float4*)(x + i0);
    ushort4 o;
    o.x = f2bf(v.x); o.y = f2bf(v.y); o.z = f2bf(v.z); o.w = f2bf(v.w);
    *(ushort4*)(xb + i0) = o;
}

// ---------------------------------------------------------------------------
// Kernel 3: implicit-GEMM conv via v_wmma_f32_16x16x32_bf16, double-buffered
// software pipeline on the CDNA5 async-to-LDS path.
// Block = 256 thr = 8 waves -> 128(M) x 128(N) tile; wave = 32x64 register
// tile (2 A-frags x 4 B-frags -> 8 WMMA / K-step). 72 K-chunks of 32
// (fixed filter tap, contiguous 32 input channels).
// Pipeline per iter: issue async A loads (ASYNCcnt) + gather B to regs for
// chunk i+1, run 8 WMMAs on chunk i, commit B regs to alternate LDS buffer,
// s_wait_asynccnt 0, one barrier.  grid = (25, 2, 32).
// ---------------------------------------------------------------------------
__global__ __launch_bounds__(256)
void conv_wmma_kernel(const unsigned short* __restrict__ xb,
                      const unsigned short* __restrict__ ws,
                      const float* __restrict__ bias,
                      float* __restrict__ out) {
    const int n0 = blockIdx.x * 128;
    const int m0 = blockIdx.y * 128;
    const int b  = blockIdx.z;

    const int t     = threadIdx.x;
    const int lane  = t & 31;
    const int wave  = t >> 5;       // 0..7
    const int wm    = wave & 3;     // 4 wave-rows  (32 M each)
    const int wn    = wave >> 2;    // 2 wave-cols  (64 N each)
    const int col   = lane & 15;
    const int khalf = lane >> 4;

    __shared__ unsigned short Asl[2][128][40];  // [buf][m][k] (+8 pad)
    __shared__ unsigned short Bsl[2][128][40];  // [buf][n][k] transposed patches

    v8f acc[2][4];
    {
        const v8f z = {0.f, 0.f, 0.f, 0.f, 0.f, 0.f, 0.f, 0.f};
        #pragma unroll
        for (int mi = 0; mi < 2; ++mi)
            #pragma unroll
            for (int f = 0; f < 4; ++f) acc[mi][f] = z;
    }

    unsigned short* wsb = const_cast<unsigned short*>(ws) + ((long)b * OC_ + m0) * KTOT_;
    const unsigned short* xbb = xb + (long)b * IC_ * HW_;

    // A-tile fill: 128 rows x 32 k = 512 b128 copies -> 2 per thread
    const int ar0 = t >> 2;          // row 0..63 (and +64)
    const int as0 = (t & 3) * 8;     // 8-element k segment
    // B-tile fill: thread -> (k-row 0..31, 16 consecutive n)
    const int bkk = t >> 3;
    const int bn0 = (t & 7) * 16;

    // per-thread spatial base for the B gather (independent of K chunk)
    const int nbase = n0 + bn0;
    const int oh0 = nbase / W_;
    const int ow0 = nbase - oh0 * W_;

    unsigned short breg[16];

    // ---- stage A for chunk 0 (async, ASYNCcnt-tracked) ----
#define LOAD_A_ASYNC(K0, BUF)                                                   \
    do {                                                                        \
        __builtin_amdgcn_global_load_async_to_lds_b128(                         \
            (__attribute__((address_space(1))) v4i*)(wsb + (long)ar0 * KTOT_ + (K0) + as0),        \
            (__attribute__((address_space(3))) v4i*)&Asl[BUF][ar0][as0], 0, 0); \
        __builtin_amdgcn_global_load_async_to_lds_b128(                         \
            (__attribute__((address_space(1))) v4i*)(wsb + (long)(ar0 + 64) * KTOT_ + (K0) + as0), \
            (__attribute__((address_space(3))) v4i*)&Asl[BUF][ar0 + 64][as0], 0, 0); \
    } while (0)

#define GATHER_B(K0)                                                            \
    do {                                                                        \
        const int r_   = (K0) >> 8;                                             \
        const int ic0_ = (K0) & 255;                                            \
        const int kh_ = r_ / 3, kw_ = r_ % 3;                                   \
        const unsigned short* xrow = xbb + (long)(ic0_ + bkk) * HW_;            \
        int nidx = nbase, oh = oh0, ow = ow0;                                   \
        int ih = oh + kh_ - 1;                                                  \
        _Pragma("unroll")                                                       \
        for (int j = 0; j < 16; ++j) {                                          \
            const int iw = ow + kw_ - 1;                                        \
            unsigned short v = 0;                                               \
            if (nidx < HW_ && ih >= 0 && ih < H_ && iw >= 0 && iw < W_)         \
                v = xrow[ih * W_ + iw];                                         \
            breg[j] = v;                                                        \
            ++nidx;                                                             \
            if (++ow == W_) { ow = 0; ++oh; ++ih; }                             \
        }                                                                       \
    } while (0)

#define COMMIT_B(BUF)                                                           \
    do {                                                                        \
        _Pragma("unroll")                                                       \
        for (int j = 0; j < 16; ++j) Bsl[BUF][bn0 + j][bkk] = breg[j];          \
    } while (0)

#define WAIT_ASYNC()                                                            \
    do {                                                                        \
        __builtin_amdgcn_s_wait_asynccnt(0);                                    \
    } while (0)

    // ---- prologue: fill buffer 0 with chunk 0 ----
    LOAD_A_ASYNC(0, 0);
    GATHER_B(0);
    COMMIT_B(0);
    WAIT_ASYNC();
    __syncthreads();

    int buf = 0;
    for (int k0 = 0; k0 < KTOT_; k0 += 32) {
        const int nk = k0 + 32;
        const int nbuf = buf ^ 1;
        const bool has_next = nk < KTOT_;

        // issue next chunk's loads before touching the matrix pipe
        if (has_next) {
            LOAD_A_ASYNC(nk, nbuf);
            GATHER_B(nk);
        }

        // ---- fragments from current buffer (CDNA5 16-bit lane layouts) ----
        union Frag { uint4 q[2]; v16bf v; };
        Frag fa[2], fb[4];
        const int amBase = wm * 32 + col;
        #pragma unroll
        for (int mi = 0; mi < 2; ++mi) {
            fa[mi].q[0] = *(const uint4*)&Asl[buf][amBase + mi * 16][khalf * 8];
            fa[mi].q[1] = *(const uint4*)&Asl[buf][amBase + mi * 16][16 + khalf * 8];
        }
        #pragma unroll
        for (int f = 0; f < 4; ++f) {
            const int bc = wn * 64 + f * 16 + col;
            fb[f].q[0] = *(const uint4*)&Bsl[buf][bc][khalf * 8];
            fb[f].q[1] = *(const uint4*)&Bsl[buf][bc][16 + khalf * 8];
        }

        #pragma unroll
        for (int mi = 0; mi < 2; ++mi)
            #pragma unroll
            for (int f = 0; f < 4; ++f)
                acc[mi][f] = __builtin_amdgcn_wmma_f32_16x16x32_bf16(
                    false, fa[mi].v, false, fb[f].v,
                    (short)0, acc[mi][f], false, false);

        // commit next B tile after the WMMAs have consumed their LDS reads
        if (has_next) {
            COMMIT_B(nbuf);
            WAIT_ASYNC();
        }
        __syncthreads();
        buf = nbuf;
    }

    // ---- epilogue: C/D layout VGPR rr -> M = rr + 8*khalf ----
    float* outb = out + (long)b * OC_ * HW_;
    const int mrowBase = m0 + wm * 32 + khalf * 8;
    #pragma unroll
    for (int mi = 0; mi < 2; ++mi) {
        #pragma unroll
        for (int rr = 0; rr < 8; ++rr) {
            const int m = mrowBase + mi * 16 + rr;
            const float bv = bias[m];
            #pragma unroll
            for (int f = 0; f < 4; ++f) {
                const int n = n0 + wn * 64 + f * 16 + col;
                if (n < HW_)
                    outb[(long)m * HW_ + n] = acc[mi][f][rr] + bv;
            }
        }
    }
#undef LOAD_A_ASYNC
#undef GATHER_B
#undef COMMIT_B
#undef WAIT_ASYNC
}

// ---------------------------------------------------------------------------
extern "C" void kernel_launch(void* const* d_in, const int* in_sizes, int n_in,
                              void* d_out, int out_size, void* d_ws, size_t ws_size,
                              hipStream_t stream) {
    const float* x    = (const float*)d_in[0];   // [32,256,56,56]
    const float* se   = (const float*)d_in[1];   // [32,64]
    const float* Wt   = (const float*)d_in[2];   // [589824,64]
    const float* bias = (const float*)d_in[3];   // [256]
    float* out = (float*)d_out;                  // [32,256,56,56]

    unsigned short* wsyn = (unsigned short*)d_ws;                       // 37.75 MB
    unsigned short* xbf  = (unsigned short*)((char*)d_ws +
                           (size_t)B_ * NWEIGHT_ * sizeof(unsigned short)); // 51.4 MB

    synth_w_kernel<<<NWEIGHT_ / 256, 256, 0, stream>>>(se, Wt, wsyn);

    const long nx = (long)B_ * IC_ * HW_;        // 25,690,112 (div by 1024)
    cvt_x_kernel<<<(int)(nx / 1024), 256, 0, stream>>>(x, xbf);

    dim3 grid(25, 2, 32);
    conv_wmma_kernel<<<grid, 256, 0, stream>>>(xbf, wsyn, bias, out);
}
